// DenseNGCNLayer_28664611733537
// MI455X (gfx1250) — matrix operand order repaired and checked by reading, not compiled, not run on hardware
//
#include <hip/hip_runtime.h>
#include <hip/hip_bf16.h>

// ---------------------------------------------------------------------------
// DenseNGCNLayer for MI455X (gfx1250, wave32):
//   base = X @ W            (bf16 WMMA, f32 accumulate)
//   base = SpMM(A, base) x2 (edge-parallel, f32 atomics; x fits in 192MB L2)
//   out  = LayerNorm(base + bias) * gamma + beta   (wave-per-row, shfl_xor)
// ---------------------------------------------------------------------------

#define IN_CH  512
#define OUT_CH 256
#define WSTRIDE 520   // padded LDS column stride (bf16 units): 1040B, 16B-aligned,
                      // 1040 % 256 == 16 -> staging/reads spread across 64 banks

typedef __attribute__((ext_vector_type(16))) __bf16 v16bf;
typedef __attribute__((ext_vector_type(8)))  __bf16 v8bf;
typedef __attribute__((ext_vector_type(8)))  float  v8f;

// ---------------------------------------------------------------------------
// GEMM: Y[nrows x 256] = X[nrows x 512] @ W[512 x 256]
// Block = 256 threads = 8 waves. Block tile = 128(M) x 16(N).
// Each wave: one 16x16 C tile, K swept 512 in steps of 32 via
// v_wmma_f32_16x16x32_bf16. Weight slab staged in LDS *transposed* (per-column
// contiguous K) so each B fragment is two ds_load_b128.
// ---------------------------------------------------------------------------
__global__ __launch_bounds__(256) void ngcn_gemm_wmma_bf16(
    const float* __restrict__ X, const float* __restrict__ W,
    float* __restrict__ Y, int nrows)
{
    __shared__ __bf16 sWT[16 * WSTRIDE];   // sWT[n][k] = W[k][n0+n], bf16

    const int n0 = blockIdx.y * 16;
    for (int i = threadIdx.x; i < IN_CH * 16; i += 256) {
        const int k = i >> 4;
        const int j = i & 15;
        sWT[j * WSTRIDE + k] = (__bf16)W[k * OUT_CH + n0 + j];
    }
    __syncthreads();

    const int wave = threadIdx.x >> 5;
    const int lane = threadIdx.x & 31;
    const int half = lane >> 4;          // 0: lanes 0-15, 1: lanes 16-31
    const int l15  = lane & 15;
    const int m0   = blockIdx.x * 128 + wave * 16;
    if (m0 >= nrows) return;             // wave-uniform; nrows % 16 == 0

    // A row for this lane (16-bit A layout: lanes 0-15 carry M=0..15)
    const float4* fr = (const float4*)(X + (size_t)(m0 + l15) * IN_CH);
    // B column for this lane: contiguous K run in LDS
    const __bf16* wcol = &sWT[l15 * WSTRIDE + half * 8];

    v8f acc = {};
    #pragma unroll
    for (int k0 = 0; k0 < IN_CH; k0 += 32) {
        // A fragment: K = k0 + half*8 + {0..7}, and +16 for elements 8..15
        const int kb = (k0 + half * 8) >> 2;
        const float4 f0 = fr[kb];
        const float4 f1 = fr[kb + 1];
        const float4 f2 = fr[kb + 4];
        const float4 f3 = fr[kb + 5];

        if (k0 + 64 < IN_CH)  // pull next K-chunk toward the caches
            __builtin_prefetch((const float*)fr + k0 + 64 + half * 8, 0, 1);

        v16bf a;
        a[0]  = (__bf16)f0.x; a[1]  = (__bf16)f0.y; a[2]  = (__bf16)f0.z; a[3]  = (__bf16)f0.w;
        a[4]  = (__bf16)f1.x; a[5]  = (__bf16)f1.y; a[6]  = (__bf16)f1.z; a[7]  = (__bf16)f1.w;
        a[8]  = (__bf16)f2.x; a[9]  = (__bf16)f2.y; a[10] = (__bf16)f2.z; a[11] = (__bf16)f2.w;
        a[12] = (__bf16)f3.x; a[13] = (__bf16)f3.y; a[14] = (__bf16)f3.z; a[15] = (__bf16)f3.w;

        // B fragment: K = k0 + half*8 + {0..7}, and +16 for elements 8..15, column n0+l15
        // two 16-byte runs -> two ds_load_b128
        const v8bf* wp = (const v8bf*)(wcol + k0);
        const v8bf b0 = wp[0];
        const v8bf b1 = wp[2];
        const v16bf b = __builtin_shufflevector(
            b0, b1, 0, 1, 2, 3, 4, 5, 6, 7, 8, 9, 10, 11, 12, 13, 14, 15);

        acc = __builtin_amdgcn_wmma_f32_16x16x32_bf16(
            /*neg_a=*/false, a, /*neg_b=*/false, b,
            /*c_mod=*/(short)0, acc, /*reuse_a=*/false, /*reuse_b=*/false);
    }

    // C layout: element v -> row m0 + v + half*8, col n0 + l15
    #pragma unroll
    for (int v = 0; v < 8; ++v)
        Y[(size_t)(m0 + v + half * 8) * OUT_CH + n0 + l15] = acc[v];
}

// ---------------------------------------------------------------------------
// SpMM: y[r] += val * x[c] per edge. One thread per channel, 8 edges / block.
// x (102 MB) is L2-resident; atomics resolve in L2.
// ---------------------------------------------------------------------------
__global__ __launch_bounds__(256) void ngcn_spmm_atomic(
    const int* __restrict__ rows, const int* __restrict__ cols,
    const float* __restrict__ vals, const float* __restrict__ x,
    float* __restrict__ y, int nedges)
{
    const int t  = threadIdx.x;          // channel 0..255
    const int e0 = blockIdx.x * 8;
    #pragma unroll
    for (int i = 0; i < 8; ++i) {
        const int e = e0 + i;
        if (e >= nedges) break;
        const int   r = rows[e];
        const int   c = cols[e];
        const float v = vals[e];
        const float xv = x[(size_t)c * OUT_CH + t];
        atomicAdd(y + (size_t)r * OUT_CH + t, v * xv);
    }
}

// ---------------------------------------------------------------------------
// Zero-fill (float4 granularity; counts here are multiples of 4).
// ---------------------------------------------------------------------------
__global__ __launch_bounds__(256) void ngcn_fill0(float4* __restrict__ p, size_t n4)
{
    const size_t i = (size_t)blockIdx.x * blockDim.x + threadIdx.x;
    if (i < n4) p[i] = float4{0.f, 0.f, 0.f, 0.f};
}

// ---------------------------------------------------------------------------
// out = LN(x + bias) * gamma + beta. One wave32 per row, 8 channels per lane.
// ---------------------------------------------------------------------------
__global__ __launch_bounds__(256) void ngcn_bias_layernorm(
    const float* __restrict__ X, const float* __restrict__ bias,
    const float* __restrict__ gamma, const float* __restrict__ beta,
    float* __restrict__ out, int nrows)
{
    const int row = blockIdx.x * 8 + (threadIdx.x >> 5);
    if (row >= nrows) return;            // wave-uniform
    const int lane = threadIdx.x & 31;

    const float* xr = X + (size_t)row * OUT_CH;
    float v[8];
    float s = 0.f, ss = 0.f;
    #pragma unroll
    for (int j = 0; j < 8; ++j) {
        const int ch = lane + j * 32;
        const float x = xr[ch] + bias[ch];
        v[j] = x;
        s  += x;
        ss += x * x;
    }
    #pragma unroll
    for (int off = 16; off > 0; off >>= 1) {
        s  += __shfl_xor(s,  off, 32);
        ss += __shfl_xor(ss, off, 32);
    }
    const float mean = s * (1.0f / OUT_CH);
    const float var  = ss * (1.0f / OUT_CH) - mean * mean;
    const float inv  = rsqrtf(var + 1e-5f);

    float* orow = out + (size_t)row * OUT_CH;
    #pragma unroll
    for (int j = 0; j < 8; ++j) {
        const int ch = lane + j * 32;
        orow[ch] = (v[j] - mean) * inv * gamma[ch] + beta[ch];
    }
}

// ---------------------------------------------------------------------------
// Pipeline: gemm -> ws | zero d_out | spmm ws->d_out | zero ws |
//           spmm d_out->ws | bias+LN ws->d_out
// d_out doubles as the spmm ping-pong buffer; ws needs N*256 floats.
// ---------------------------------------------------------------------------
extern "C" void kernel_launch(void* const* d_in, const int* in_sizes, int n_in,
                              void* d_out, int out_size, void* d_ws, size_t ws_size,
                              hipStream_t stream) {
    (void)n_in; (void)out_size; (void)ws_size;

    const float* features = (const float*)d_in[0];
    const int*   adj      = (const int*)  d_in[1];   // [2, E] row-major int32
    const float* vals     = (const float*)d_in[2];
    const float* weight   = (const float*)d_in[3];
    const float* bias     = (const float*)d_in[4];
    const float* gamma    = (const float*)d_in[5];
    const float* beta     = (const float*)d_in[6];

    const int nrows  = in_sizes[0] / IN_CH;
    const int nedges = in_sizes[2];

    float* bufA = (float*)d_ws;
    float* outp = (float*)d_out;

    const size_t total = (size_t)nrows * OUT_CH;
    const size_t n4    = total / 4;
    const int fill_blocks = (int)((n4 + 255) / 256);
    const int spmm_blocks = (nedges + 7) / 8;

    // 1) base = X @ W  -> ws
    dim3 ggrid((nrows + 127) / 128, OUT_CH / 16);
    ngcn_gemm_wmma_bf16<<<ggrid, 256, 0, stream>>>(features, weight, bufA, nrows);

    // 2) base = SpMM(base)  (ws -> d_out)
    ngcn_fill0<<<fill_blocks, 256, 0, stream>>>((float4*)outp, n4);
    ngcn_spmm_atomic<<<spmm_blocks, 256, 0, stream>>>(adj, adj + nedges, vals,
                                                      bufA, outp, nedges);

    // 3) base = SpMM(base)  (d_out -> ws)
    ngcn_fill0<<<fill_blocks, 256, 0, stream>>>((float4*)bufA, n4);
    ngcn_spmm_atomic<<<spmm_blocks, 256, 0, stream>>>(adj, adj + nedges, vals,
                                                      outp, bufA, nedges);

    // 4) out = LN(base + bias) * gamma + beta  (ws -> d_out)
    ngcn_bias_layernorm<<<(nrows + 7) / 8, 256, 0, stream>>>(bufA, bias, gamma,
                                                             beta, outp, nrows);
}
